// GRUModel_66778151518945
// MI455X (gfx1250) — matrix operand order repaired and checked by reading, not compiled, not run on hardware
//
#include <hip/hip_runtime.h>
#include <hip/hip_bf16.h>
#include <math.h>

#define NEGV (-1000000000.0f)
#define Bb   64
#define Ss   128
#define Ee   512
#define Hh   512
#define NPOI 8192
#define NCAT 400
#define NCATP 448          // NCAT padded to a multiple of 64 for the unconditional GEMM
#define NFf  128

typedef __attribute__((ext_vector_type(16))) __bf16 v16bf;
typedef __attribute__((ext_vector_type(8)))  __bf16 v8bf;
typedef __attribute__((ext_vector_type(8)))  float  v8f;

// ---------------- WMMA fragment helpers (pre-converted bf16 operands) -------
// A-matrix 16x32 bf16 (ISA 7.12.2): lane row M=lane%16; lo lanes K {k0..k0+7, k0+16..k0+23},
// hi lanes shifted by +8. Each 8-element run is 16B -> one global_load_b128, already packed.
__device__ __forceinline__ v16bf load_frag_a(const __bf16* __restrict__ rowptr, int k0, int lane) {
  const __bf16* base = rowptr + k0 + ((lane & 16) ? 8 : 0);
  v8bf lo = *(const v8bf*)(base);
  v8bf hi = *(const v8bf*)(base + 16);
  return __builtin_shufflevector(lo, hi, 0, 1, 2, 3, 4, 5, 6, 7, 8, 9, 10, 11, 12, 13, 14, 15);
}
// B-matrix 32x16 bf16: lane col N=lane%16; lo lanes K {k0..k0+7, k0+16..k0+23}, hi lanes shifted by +8
// (same K order as A). Each 8-element run is 16B aligned -> two vector loads.
__device__ __forceinline__ v16bf load_frag_b(const __bf16* __restrict__ rowptr, int k0, int lane) {
  const __bf16* base = rowptr + k0 + ((lane & 16) ? 8 : 0);
  v8bf lo = *(const v8bf*)(base);
  v8bf hi = *(const v8bf*)(base + 16);
  return __builtin_shufflevector(lo, hi, 0, 1, 2, 3, 4, 5, 6, 7, 8, 9, 10, 11, 12, 13, 14, 15);
}
__device__ __forceinline__ v8f acc_zero() { v8f a = {}; return a; }

__device__ __forceinline__ float sigm(float x) { return 1.0f / (1.0f + expf(-x)); }

// ---------------- fp32 -> bf16 pre-conversion (one streaming pass) ----------
__global__ void cvt_f32_bf16(const float* __restrict__ in, __bf16* __restrict__ out, int n) {
  int i = (blockIdx.x * blockDim.x + threadIdx.x) * 4;
  if (i >= n) return;
#pragma unroll
  for (int t = 0; t < 4; ++t) out[i + t] = (__bf16)in[i + t];
}

// W_cat (400x512) -> bf16 padded to 448x512 with zero rows (keeps GEMM unconditional).
__global__ void cvt_pad_cat(const float* __restrict__ in, __bf16* __restrict__ out) {
  int i = (blockIdx.x * blockDim.x + threadIdx.x) * 4;
  if (i >= NCATP * Hh) return;
  int row = i / Hh;
#pragma unroll
  for (int t = 0; t < 4; ++t)
    out[i + t] = (row < NCAT) ? (__bf16)in[i + t] : (__bf16)0.0f;
}

// ---------------- Big GEMM: C(MxN) = A(MxK) @ W(NxK)^T (+bias), bf16 in -----
// Per-wave 64x64 tile: 4 M-subtiles x 4 N-subtiles, K-loop of 16x16x32 bf16 WMMA.
// Requires M % 64 == 0, N % 64 == 0, K % 32 == 0 (guaranteed by padding).
__global__ void __launch_bounds__(256)
gemm_xwt_bf16(const __bf16* __restrict__ A, const __bf16* __restrict__ W,
              const float* __restrict__ bias, float* __restrict__ C,
              int M, int N, int K) {
  const int lane = threadIdx.x & 31;
  const int wid  = (int)((blockIdx.x * blockDim.x + threadIdx.x) >> 5);
  const int ntn  = N >> 6;          // 64-wide col blocks
  const int ntm  = M >> 6;          // 64-tall row blocks
  if (wid >= ntn * ntm) return;     // wave-uniform
  const int wm = wid / ntn, wn = wid % ntn;
  const int row0 = wm << 6, col0 = wn << 6;
  const int r = lane & 15;

  const __bf16* arow[4];
  const __bf16* brow[4];
#pragma unroll
  for (int i = 0; i < 4; ++i) {
    arow[i] = A + (size_t)(row0 + 16 * i + r) * K;
    brow[i] = W + (size_t)(col0 + 16 * i + r) * K;
  }

  v8f acc[4][4];
#pragma unroll
  for (int i = 0; i < 4; ++i)
#pragma unroll
    for (int j = 0; j < 4; ++j) acc[i][j] = acc_zero();

  for (int k0 = 0; k0 < K; k0 += 32) {
    v16bf a[4];
#pragma unroll
    for (int i = 0; i < 4; ++i) a[i] = load_frag_a(arow[i], k0, lane);
#pragma unroll
    for (int j = 0; j < 4; ++j) {
      v16bf b = load_frag_b(brow[j], k0, lane);
#pragma unroll
      for (int i = 0; i < 4; ++i)
        acc[i][j] = __builtin_amdgcn_wmma_f32_16x16x32_bf16(false, a[i], false, b, (short)0, acc[i][j], false, false);
    }
  }

  const int mb = (lane & 16) ? 8 : 0;
#pragma unroll
  for (int i = 0; i < 4; ++i)
#pragma unroll
    for (int j = 0; j < 4; ++j) {
      int n = col0 + 16 * j + r;
      float bv = bias ? bias[n] : 0.0f;
#pragma unroll
      for (int t = 0; t < 8; ++t)
        C[(size_t)(row0 + 16 * i + mb + t) * N + n] = acc[i][j][t] + bv;
    }
}

// ---------------- Node attention precompute ---------------------------------
// c1 = W_attn @ a[:NA], c2 = W_attn @ a[NA:]  (collapses Wh1/Wh2 to mat-vec over X)
__global__ void attn_c_kernel(const float* __restrict__ W_attn,
                              const float* __restrict__ a_attn, float* __restrict__ c) {
  int k = threadIdx.x;  // 128
  float s1 = 0.f, s2 = 0.f;
  for (int j = 0; j < NFf; ++j) {
    float w = W_attn[k * NFf + j];
    s1 += w * a_attn[j];
    s2 += w * a_attn[NFf + j];
  }
  c[k] = s1;
  c[NFf + k] = s2;
}

__global__ void attn_wh_kernel(const float* __restrict__ X, const float* __restrict__ c,
                               float* __restrict__ wh1, float* __restrict__ wh2) {
  int p = blockIdx.x * blockDim.x + threadIdx.x;
  if (p >= NPOI) return;
  const float* row = X + (size_t)p * NFf;
  float s1 = 0.f, s2 = 0.f;
  for (int k = 0; k < NFf; ++k) { s1 += row[k] * c[k]; s2 += row[k] * c[NFf + k]; }
  wh1[p] = s1;
  wh2[p] = s2;
}

// Per (b,j): masked max/argmax over k<j (gather of A), then softmax([v0,v1]).
__global__ void attn_stats_kernel(const float* __restrict__ wh1, const float* __restrict__ wh2,
                                  const float* __restrict__ Amat, const int* __restrict__ seqs,
                                  const int* __restrict__ lens, float* __restrict__ v,
                                  int* __restrict__ indices) {
  const int lane = threadIdx.x & 31;
  const int wid  = (int)((blockIdx.x * blockDim.x + threadIdx.x) >> 5);  // 0..B*S-1
  const int b = wid >> 7, j = wid & 127;
  const int len = lens[b];
  const int qj = seqs[b * Ss + j];
  const float whj = wh2[qj];
  const bool jvalid = (j < len);

  float best = NEGV;
  int bidx = 0;
  for (int k = lane; k < j; k += 32) {
    float val = NEGV;
    if (jvalid) {
      int qk = seqs[b * Ss + k];
      float e = wh1[qk] + whj;
      e = (e > 0.f) ? e : 0.2f * e;                       // leaky_relu 0.2
      val = e * (Amat[(size_t)qk * NPOI + qj] + 1.0f);
    }
    if (val > best) { best = val; bidx = k; }             // first-max per lane (ascending k)
  }
#pragma unroll
  for (int off = 16; off > 0; off >>= 1) {
    float ov = __shfl_xor(best, off, 32);
    int   oi = __shfl_xor(bidx, off, 32);
    if (ov > best || (ov == best && oi < bidx)) { best = ov; bidx = oi; }
  }
  if (lane == 0) {
    float v0 = best, v1;
    if (j == 0)        v1 = 0.0f;
    else if (!jvalid)  v1 = NEGV;
    else {
      int qk = seqs[b * Ss + j - 1];
      float e = wh1[qk] + whj;
      e = (e > 0.f) ? e : 0.2f * e;
      v1 = e * (Amat[(size_t)qk * NPOI + qj] + 1.0f);
    }
    float m  = fmaxf(v0, v1);
    float e0 = expf(v0 - m), e1 = expf(v1 - m);
    float inv = 1.0f / (e0 + e1);
    v[wid * 2 + 0] = e0 * inv;
    v[wid * 2 + 1] = e1 * inv;
    indices[wid] = bidx;
  }
}

// ---------------- Sequential GRU scan ---------------------------------------
// Phase A: gh1 = h1 @ W_hh^T, gh2 = h2 @ W_hh^T  (M=64, N=1536, K=512),
// h1 = xbuf[:,step-1], h2 = xbuf[b, indices[b,step]]; step 0 uses broadcast h0.
// 768 waves, each one 16x16 tile via bf16 WMMA. Gathered A rows via per-lane pointers.
__global__ void __launch_bounds__(256)
gru_hgemm_step(const __bf16* __restrict__ xbuf_bf, const __bf16* __restrict__ h0_bf,
               const int* __restrict__ indices, const __bf16* __restrict__ W_hh_bf,
               float* __restrict__ gh1, float* __restrict__ gh2, int step) {
  const int lane = threadIdx.x & 31;
  int wid = (int)((blockIdx.x * blockDim.x + threadIdx.x) >> 5);  // 0..767
  const int which = wid / 384;  // 0 -> gh1, 1 -> gh2
  wid -= which * 384;
  const int mt = wid / 96, nt = wid % 96;
  const int r = lane & 15;
  const int b = mt * 16 + r;

  const __bf16* hrow;
  if (step == 0)          hrow = h0_bf;                                           // broadcast (1,H)
  else if (which == 0)    hrow = xbuf_bf + ((size_t)b * Ss + (step - 1)) * Hh;
  else                    hrow = xbuf_bf + ((size_t)b * Ss + indices[b * Ss + step]) * Hh;
  const __bf16* wrow = W_hh_bf + (size_t)(nt * 16 + r) * Hh;

  v8f acc = acc_zero();
  for (int k0 = 0; k0 < Hh; k0 += 32) {
    v16bf a = load_frag_a(hrow, k0, lane);
    v16bf w = load_frag_b(wrow, k0, lane);
    acc = __builtin_amdgcn_wmma_f32_16x16x32_bf16(false, a, false, w, (short)0, acc, false, false);
  }

  float* out = (which == 0) ? gh1 : gh2;
  const int mb = (lane & 16) ? 8 : 0;
  const int n = nt * 16 + r;
#pragma unroll
  for (int t = 0; t < 8; ++t)
    out[(size_t)(mt * 16 + mb + t) * (3 * Hh) + n] = acc[t];
}

// Phase B: gate math + blend; writes xbuf[:, step] (fp32) and its bf16 mirror.
__global__ void gru_gate_step(const float* __restrict__ gi, const float* __restrict__ gh1,
                              const float* __restrict__ gh2, const float* __restrict__ b_ih,
                              const float* __restrict__ b_hh, const float* __restrict__ h0,
                              const int* __restrict__ indices, const float* __restrict__ v,
                              float* __restrict__ xbuf, __bf16* __restrict__ xbuf_bf, int step) {
  int idx = blockIdx.x * blockDim.x + threadIdx.x;
  if (idx >= Bb * Hh) return;
  int b = idx >> 9, h = idx & (Hh - 1);

  size_t gir = ((size_t)b * Ss + step) * (3 * Hh);
  float ir = gi[gir + h]           + b_ih[h];
  float iz = gi[gir + Hh + h]      + b_ih[Hh + h];
  float in_ = gi[gir + 2 * Hh + h] + b_ih[2 * Hh + h];

  size_t ghr = (size_t)b * (3 * Hh);
  float hr1 = gh1[ghr + h]          + b_hh[h];
  float hz1 = gh1[ghr + Hh + h]     + b_hh[Hh + h];
  float hn1 = gh1[ghr + 2 * Hh + h] + b_hh[2 * Hh + h];
  float hr2 = gh2[ghr + h]          + b_hh[h];
  float hz2 = gh2[ghr + Hh + h]     + b_hh[Hh + h];
  float hn2 = gh2[ghr + 2 * Hh + h] + b_hh[2 * Hh + h];

  float h1v, h2v, a0, a1;
  if (step == 0) {
    h1v = h0[h]; h2v = h0[h]; a0 = 1.0f; a1 = 0.0f;      // first step uses hid1 only
  } else {
    h1v = xbuf[((size_t)b * Ss + (step - 1)) * Hh + h];
    h2v = xbuf[((size_t)b * Ss + indices[b * Ss + step]) * Hh + h];
    a0 = v[(b * Ss + step) * 2 + 0];
    a1 = v[(b * Ss + step) * 2 + 1];
  }

  float r1 = sigm(ir + hr1), z1 = sigm(iz + hz1);
  float n1 = tanhf(in_ + r1 * hn1);
  float hid1 = (1.0f - z1) * n1 + z1 * h1v;

  float r2 = sigm(ir + hr2), z2 = sigm(iz + hz2);
  float n2 = tanhf(in_ + r2 * hn2);
  float hid2 = (1.0f - z2) * n2 + z2 * h2v;

  float hn = a0 * hid1 + a1 * hid2;
  size_t oidx = ((size_t)b * Ss + step) * Hh + h;
  xbuf[oidx] = hn;
  xbuf_bf[oidx] = (__bf16)hn;
}

// ---------------- out_time (rank-1 head) ------------------------------------
__global__ void time_proj(const float* __restrict__ xbuf, const float* __restrict__ W_time,
                          const float* __restrict__ b_time, float* __restrict__ out) {
  const int lane = threadIdx.x & 31;
  const int wid  = (int)((blockIdx.x * blockDim.x + threadIdx.x) >> 5);  // 0..B*S-1
  if (wid >= Bb * Ss) return;
  const float* row = xbuf + (size_t)wid * Hh;
  float s = 0.f;
  for (int k = lane; k < Hh; k += 32) s += row[k] * W_time[k];
#pragma unroll
  for (int off = 16; off > 0; off >>= 1) s += __shfl_xor(s, off, 32);
  if (lane == 0) out[wid] = s + b_time[0];
}

// Depad out_cat (8192x448 -> 8192x400) and add bias.
__global__ void cat_depad_bias(const float* __restrict__ pad, const float* __restrict__ b_cat,
                               float* __restrict__ out) {
  int idx = blockIdx.x * blockDim.x + threadIdx.x;
  if (idx >= Bb * Ss * NCAT) return;
  int m = idx / NCAT, n = idx - m * NCAT;
  out[idx] = pad[(size_t)m * NCATP + n] + b_cat[n];
}

// ---------------- launch ----------------------------------------------------
extern "C" void kernel_launch(void* const* d_in, const int* in_sizes, int n_in,
                              void* d_out, int out_size, void* d_ws, size_t ws_size,
                              hipStream_t stream) {
  (void)in_sizes; (void)n_in; (void)out_size; (void)ws_size;
  const float* src    = (const float*)d_in[0];
  const int*   lens   = (const int*)  d_in[1];
  const int*   seqs   = (const int*)  d_in[2];
  const float* X      = (const float*)d_in[3];
  const float* Amat   = (const float*)d_in[4];
  const float* W_attn = (const float*)d_in[5];
  const float* a_attn = (const float*)d_in[6];
  const float* W_ih   = (const float*)d_in[7];
  const float* W_hh   = (const float*)d_in[8];
  const float* b_ih   = (const float*)d_in[9];
  const float* b_hh   = (const float*)d_in[10];
  const float* h0     = (const float*)d_in[11];
  const float* W_poi  = (const float*)d_in[12];
  const float* b_poi  = (const float*)d_in[13];
  const float* W_time = (const float*)d_in[14];
  const float* b_time = (const float*)d_in[15];
  const float* W_cat  = (const float*)d_in[16];
  const float* b_cat  = (const float*)d_in[17];

  // ---- workspace layout: fp32 region, then bf16 region (32B-aligned) ----
  float* ws   = (float*)d_ws;
  float* c    = ws;                                // 256
  float* wh1  = c + 256;                           // 8192
  float* wh2  = wh1 + NPOI;                        // 8192
  float* vsm  = wh2 + NPOI;                        // B*S*2 = 16384
  int* indices = (int*)(vsm + Bb * Ss * 2);        // 8192 ints
  float* gh1  = (float*)(indices + Bb * Ss);       // 64*1536
  float* gh2  = gh1 + Bb * 3 * Hh;                 // 64*1536
  float* gi   = gh2 + Bb * 3 * Hh;                 // 8192*1536
  float* xbuf = gi + (size_t)Bb * Ss * 3 * Hh;     // 8192*512
  float* cat_pad = xbuf + (size_t)Bb * Ss * Hh;    // 8192*448
  __bf16* bfp    = (__bf16*)(cat_pad + (size_t)Bb * Ss * NCATP);
  __bf16* src_bf  = bfp;                               // 8192*512
  __bf16* wih_bf  = src_bf  + (size_t)Bb * Ss * Ee;    // 1536*512
  __bf16* whh_bf  = wih_bf  + (size_t)3 * Hh * Ee;     // 1536*512
  __bf16* wpoi_bf = whh_bf  + (size_t)3 * Hh * Hh;     // 8192*512
  __bf16* wcat_bf = wpoi_bf + (size_t)NPOI * Hh;       // 448*512 (padded)
  __bf16* h0_bf   = wcat_bf + (size_t)NCATP * Hh;      // 512
  __bf16* xbuf_bf = h0_bf + Hh;                        // 8192*512

  float* out_poi  = (float*)d_out;
  float* out_time = out_poi + (size_t)Bb * Ss * NPOI;
  float* out_cat  = out_time + Bb * Ss;

  // one-time fp32 -> bf16 operand conversion (small streaming pass, ~60MB)
  cvt_f32_bf16<<<(Bb * Ss * Ee / 4 + 255) / 256, 256, 0, stream>>>(src, src_bf, Bb * Ss * Ee);
  cvt_f32_bf16<<<(3 * Hh * Ee / 4 + 255) / 256, 256, 0, stream>>>(W_ih, wih_bf, 3 * Hh * Ee);
  cvt_f32_bf16<<<(3 * Hh * Hh / 4 + 255) / 256, 256, 0, stream>>>(W_hh, whh_bf, 3 * Hh * Hh);
  cvt_f32_bf16<<<(NPOI * Hh / 4 + 255) / 256, 256, 0, stream>>>(W_poi, wpoi_bf, NPOI * Hh);
  cvt_pad_cat<<<(NCATP * Hh / 4 + 255) / 256, 256, 0, stream>>>(W_cat, wcat_bf);
  cvt_f32_bf16<<<1, 128, 0, stream>>>(h0, h0_bf, Hh);

  // node attention (rank-1 collapse + gather stats)
  attn_c_kernel<<<1, 128, 0, stream>>>(W_attn, a_attn, c);
  attn_wh_kernel<<<NPOI / 256, 256, 0, stream>>>(X, c, wh1, wh2);
  attn_stats_kernel<<<(Bb * Ss * 32) / 256, 256, 0, stream>>>(wh1, wh2, Amat, seqs, lens, vsm, indices);

  // gi = src @ W_ih^T  (M=8192, N=1536, K=512): (128*24) waves -> 384 blocks
  gemm_xwt_bf16<<<384, 256, 0, stream>>>(src_bf, wih_bf, nullptr, gi, Bb * Ss, 3 * Hh, Ee);

  // sequential GRU scan (state lives in xbuf / xbuf_bf)
  for (int step = 0; step < Ss; ++step) {
    gru_hgemm_step<<<96, 256, 0, stream>>>(xbuf_bf, h0_bf, indices, whh_bf, gh1, gh2, step);
    gru_gate_step<<<(Bb * Hh) / 256, 256, 0, stream>>>(gi, gh1, gh2, b_ih, b_hh, h0, indices, vsm,
                                                       xbuf, xbuf_bf, step);
  }

  // heads
  gemm_xwt_bf16<<<2048, 256, 0, stream>>>(xbuf_bf, wpoi_bf, b_poi, out_poi, Bb * Ss, NPOI, Hh);   // 128*128 waves
  gemm_xwt_bf16<<<112, 256, 0, stream>>>(xbuf_bf, wcat_bf, nullptr, cat_pad, Bb * Ss, NCATP, Hh); // 128*7 waves
  cat_depad_bias<<<(Bb * Ss * NCAT + 255) / 256, 256, 0, stream>>>(cat_pad, b_cat, out_cat);
  time_proj<<<1024, 256, 0, stream>>>(xbuf, W_time, b_time, out_time);
}
